// TransformerClassification_32332513804869
// MI455X (gfx1250) — compile-verified, matrix operand order checked by reading, not run on hardware
//
#include <hip/hip_runtime.h>

#define IN_F  128
#define HCC   64
#define HEADS 4
#define CH    16
#define OUTF  7

typedef __attribute__((ext_vector_type(16))) _Float16 v16h;
typedef __attribute__((ext_vector_type(8)))  _Float16 v8h;
typedef __attribute__((ext_vector_type(8)))  float    v8f;

// ---------------------------------------------------------------------------
// Float atomic-max via CAS (portable, graph-capture safe)
// ---------------------------------------------------------------------------
__device__ __forceinline__ void atomicMaxF(float* addr, float val) {
    unsigned int* p = (unsigned int*)addr;
    unsigned int old = *p;
    while (__uint_as_float(old) < val) {
        unsigned int assumed = old;
        old = atomicCAS(p, assumed, __float_as_uint(val));
        if (old == assumed) break;
    }
}

// ---------------------------------------------------------------------------
// Workspace init: amax = -1e30, denom = 0, accumulators = 0 (every call)
// ---------------------------------------------------------------------------
__global__ void init_ws_kernel(float* amax1, float* den1, float* acc1,
                               float* amax2, float* den2, float* acc2, int N) {
    long i = (long)blockIdx.x * blockDim.x + threadIdx.x;
    if (i < (long)N * HCC) acc1[i] = 0.f;
    if (i < (long)N * HEADS) { amax1[i] = -1e30f; den1[i] = 0.f; }
    if (i < (long)N * OUTF) acc2[i] = 0.f;
    if (i < N) { amax2[i] = -1e30f; den2[i] = 0.f; }
}

// ---------------------------------------------------------------------------
// Weight pre-pack, conv1: f32 [128 x 64] -> f16 WMMA B-fragments.
// Layout: pack1[m*8192 + ct*2048 + kk*512 + lane*16 + j]
//   (m = matrix 0..3, ct = 16-col tile 0..3, kk = K-chunk of 32, 0..3)
// B element (lane, j): K = kk*32 + (lane>=16 ? 8:0) + (j<8 ? j : j+8),
//                      col = ct*16 + (lane & 15)      (ISA 7.12.2 layout)
// ---------------------------------------------------------------------------
__global__ void pack_w1_kernel(const float* __restrict__ Wq,
                               const float* __restrict__ Wk,
                               const float* __restrict__ Wv,
                               const float* __restrict__ Ws,
                               _Float16* __restrict__ pack) {
    const int m = blockIdx.y;                       // uniform matrix select
    const float* W = (m == 0) ? Wq : (m == 1) ? Wk : (m == 2) ? Wv : Ws;
    int t = blockIdx.x * blockDim.x + threadIdx.x;  // [0, 8192)
    const int j    = t & 15;
    const int lane = (t >> 4) & 31;
    const int kk   = (t >> 9) & 3;
    const int ct   = (t >> 11) & 3;
    const int K    = kk * 32 + ((lane & 16) ? 8 : 0) + ((j < 8) ? j : j + 8);
    const int col  = ct * 16 + (lane & 15);
    pack[(size_t)m * 8192 + t] = (_Float16)W[K * HCC + col];
}

// ---------------------------------------------------------------------------
// Weight pre-pack, conv2: f32 [64 x 7] -> f16 B-fragments, cols>=7 zero.
// Layout: pack2[m*1024 + kk*512 + lane*16 + j]   (kk = 0..1)
// ---------------------------------------------------------------------------
__global__ void pack_w2_kernel(const float* __restrict__ Wq,
                               const float* __restrict__ Wk,
                               const float* __restrict__ Wv,
                               const float* __restrict__ Ws,
                               _Float16* __restrict__ pack) {
    const int m = blockIdx.y;
    const float* W = (m == 0) ? Wq : (m == 1) ? Wk : (m == 2) ? Wv : Ws;
    int t = blockIdx.x * blockDim.x + threadIdx.x;  // [0, 1024)
    const int j    = t & 15;
    const int lane = (t >> 4) & 31;
    const int kk   = (t >> 9) & 1;
    const int K    = kk * 32 + ((lane & 16) ? 8 : 0) + ((j < 8) ? j : j + 8);
    const int col  = lane & 15;
    pack[(size_t)m * 1024 + t] =
        (col < OUTF) ? (_Float16)W[K * OUTF + col] : (_Float16)0.f;
}

// ---------------------------------------------------------------------------
// conv1 GEMMs: q/k/v/s = x @ W + b    [N x 128] @ [128 x 64]
// One wave per 16x16 tile; B-fragments pre-packed, A staged in LDS as f16.
// ---------------------------------------------------------------------------
__global__ void __launch_bounds__(256)
gemm_conv1_kernel(const float* __restrict__ x,
                  const _Float16* __restrict__ pack,
                  const float* __restrict__ bq, const float* __restrict__ bk,
                  const float* __restrict__ bv, const float* __restrict__ bs,
                  float* __restrict__ q, float* __restrict__ k,
                  float* __restrict__ v, float* __restrict__ s, int N) {
    __shared__ __align__(16) _Float16 xt[16 * IN_F];    // 4 KB f16 x-tile
    const int tid  = threadIdx.x;
    const int row0 = blockIdx.x * 16;

    {   // stage: each thread converts 8 consecutive floats of one row
        const int idx0 = tid * 8;
        const int r = idx0 >> 7, c = idx0 & 127;
        int gr = row0 + r; if (gr >= N) gr = N - 1;
        const float* src = x + (long)gr * IN_F + c;
        v8h hv;
#pragma unroll
        for (int i = 0; i < 8; ++i) hv[i] = (_Float16)src[i];
        *(v8h*)(xt + idx0) = hv;
    }
    __syncthreads();

    const int wave = tid >> 5;
    const int lane = tid & 31;
    const int hi   = lane >> 4;
    const int base = hi * 8;
    const int lrow = lane & 15;
    const int ct   = wave & 3;
    const int col  = ct * 16 + lrow;
    const bool fullTile = (row0 + 16 <= N);
    const _Float16* xrow = xt + lrow * IN_F;

    auto run_tile = [&](const _Float16* __restrict__ Wp,
                        const float* __restrict__ bias,
                        float* __restrict__ O) {
        v8f acc = {};
#pragma unroll
        for (int kkI = 0; kkI < 4; ++kkI) {
            const _Float16* ap = xrow + kkI * 32 + base;
            v8h alo = *(const v8h*)ap;
            v8h ahi = *(const v8h*)(ap + 16);
            v16h a = __builtin_shufflevector(alo, ahi,
                        0, 1, 2, 3, 4, 5, 6, 7, 8, 9, 10, 11, 12, 13, 14, 15);
            v16h b = *(const v16h*)(Wp + (size_t)kkI * 512 + lane * 16);
            acc = __builtin_amdgcn_wmma_f32_16x16x32_f16(
                      false, a, false, b, (short)0, acc, false, false);
        }
        const float bb = bias[col];
        float* Ob = O + (long)(row0 + hi * 8) * HCC + col;
        if (fullTile) {
#pragma unroll
            for (int i = 0; i < 8; ++i) Ob[(long)i * HCC] = acc[i] + bb;
        } else {
#pragma unroll
            for (int i = 0; i < 8; ++i)
                if (row0 + hi * 8 + i < N) Ob[(long)i * HCC] = acc[i] + bb;
        }
    };

    // waves 0-3: matrices 0 (q) then 2 (v); waves 4-7: matrices 1 (k) then 3 (s)
    if (wave < 4) {
        run_tile(pack + (size_t)(0 * 8192 + ct * 2048), bq, q);
        run_tile(pack + (size_t)(2 * 8192 + ct * 2048), bv, v);
    } else {
        run_tile(pack + (size_t)(1 * 8192 + ct * 2048), bk, k);
        run_tile(pack + (size_t)(3 * 8192 + ct * 2048), bs, s);
    }
}

// ---------------------------------------------------------------------------
// conv2 GEMMs: q2/k2/v2/s2 = h @ W + b    [N x 64] @ [64 x 7]
// ---------------------------------------------------------------------------
__global__ void __launch_bounds__(256)
gemm_conv2_kernel(const float* __restrict__ h,
                  const _Float16* __restrict__ pack,
                  const float* __restrict__ bq, const float* __restrict__ bk,
                  const float* __restrict__ bv, const float* __restrict__ bs,
                  float* __restrict__ q, float* __restrict__ k,
                  float* __restrict__ v, float* __restrict__ s, int N) {
    __shared__ __align__(16) _Float16 ht[16 * HCC];     // 2 KB f16 h-tile
    const int tid  = threadIdx.x;
    const int row0 = blockIdx.x * 16;

    {   // 1024 elems / 256 threads: 4 consecutive floats each
        const int idx0 = tid * 4;
        const int r = idx0 >> 6, c = idx0 & 63;
        int gr = row0 + r; if (gr >= N) gr = N - 1;
        const float* src = h + (long)gr * HCC + c;
        _Float16 h0 = (_Float16)src[0], h1 = (_Float16)src[1];
        _Float16 h2 = (_Float16)src[2], h3 = (_Float16)src[3];
        ht[idx0] = h0; ht[idx0 + 1] = h1; ht[idx0 + 2] = h2; ht[idx0 + 3] = h3;
    }
    __syncthreads();

    const int wave = tid >> 5;
    if (wave >= 4) return;
    const int lane = tid & 31;
    const int hi   = lane >> 4;
    const int base = hi * 8;
    const int lrow = lane & 15;
    const int col  = lrow;
    const _Float16* hrow = ht + lrow * HCC;

    auto run_tile = [&](const _Float16* __restrict__ Wp,
                        const float* __restrict__ bias,
                        float* __restrict__ O) {
        v8f acc = {};
#pragma unroll
        for (int kkI = 0; kkI < 2; ++kkI) {
            const _Float16* ap = hrow + kkI * 32 + base;
            v8h alo = *(const v8h*)ap;
            v8h ahi = *(const v8h*)(ap + 16);
            v16h a = __builtin_shufflevector(alo, ahi,
                        0, 1, 2, 3, 4, 5, 6, 7, 8, 9, 10, 11, 12, 13, 14, 15);
            v16h b = *(const v16h*)(Wp + (size_t)kkI * 512 + lane * 16);
            acc = __builtin_amdgcn_wmma_f32_16x16x32_f16(
                      false, a, false, b, (short)0, acc, false, false);
        }
        if (col < OUTF) {
            const float bb = bias[col];
#pragma unroll
            for (int i = 0; i < 8; ++i) {
                int gr = row0 + hi * 8 + i;
                if (gr < N) O[(long)gr * OUTF + col] = acc[i] + bb;
            }
        }
    };

    if (wave == 0)      run_tile(pack,        bq, q);
    else if (wave == 1) run_tile(pack + 1024, bk, k);
    else if (wave == 2) run_tile(pack + 2048, bv, v);
    else                run_tile(pack + 3072, bs, s);
}

// ---------------------------------------------------------------------------
// conv1 edge pass 1: logits + segment max  (alpha stored in att1)
// ---------------------------------------------------------------------------
__global__ void edge1_logits_kernel(const float* __restrict__ q,
                                    const float* __restrict__ k,
                                    const float* __restrict__ ea,
                                    const float* __restrict__ We,
                                    const int* __restrict__ ei, int E,
                                    float* __restrict__ att,
                                    float* __restrict__ amax) {
    __shared__ float Wes[2 * HCC];
    if (threadIdx.x < 2 * HCC) Wes[threadIdx.x] = We[threadIdx.x];
    __syncthreads();
    long e = (long)blockIdx.x * blockDim.x + threadIdx.x;
    if (e >= E) return;
    const int src = ei[e], dst = ei[(long)E + e];
    const float a0 = ea[e * 2], a1 = ea[e * 2 + 1];
    const float* qd = q + (long)dst * HCC;
    const float* ks = k + (long)src * HCC;
    __builtin_prefetch(ks, 0, 0);
#pragma unroll
    for (int hh = 0; hh < HEADS; ++hh) {
        float acc = 0.f;
#pragma unroll
        for (int c = 0; c < CH; ++c) {
            const int hc = hh * CH + c;
            const float ev = a0 * Wes[hc] + a1 * Wes[HCC + hc];
            acc += qd[hc] * (ks[hc] + ev);
        }
        acc *= 0.25f;                           // 1/sqrt(16)
        att[e * HEADS + hh] = acc;
        atomicMaxF(&amax[(long)dst * HEADS + hh], acc);
    }
}

// ---------------------------------------------------------------------------
// conv1 edge pass 2: exp + segment sum   (att: alpha -> exp in place)
// ---------------------------------------------------------------------------
__global__ void edge1_exp_kernel(float* __restrict__ att,
                                 const float* __restrict__ amax,
                                 const int* __restrict__ ei, int E,
                                 float* __restrict__ den) {
    long e = (long)blockIdx.x * blockDim.x + threadIdx.x;
    if (e >= E) return;
    const int dst = ei[(long)E + e];
#pragma unroll
    for (int hh = 0; hh < HEADS; ++hh) {
        const float ex = __expf(att[e * HEADS + hh] - amax[(long)dst * HEADS + hh]);
        att[e * HEADS + hh] = ex;
        atomicAdd(&den[(long)dst * HEADS + hh], ex);
    }
}

// ---------------------------------------------------------------------------
// conv1 edge pass 3: weighted scatter-add of (v[src] + e)
// ---------------------------------------------------------------------------
__global__ void edge1_accum_kernel(const float* __restrict__ v,
                                   const float* __restrict__ ea,
                                   const float* __restrict__ We,
                                   const float* __restrict__ att,
                                   const float* __restrict__ den,
                                   const int* __restrict__ ei, int E,
                                   float* __restrict__ acc) {
    __shared__ float Wes[2 * HCC];
    if (threadIdx.x < 2 * HCC) Wes[threadIdx.x] = We[threadIdx.x];
    __syncthreads();
    long e = (long)blockIdx.x * blockDim.x + threadIdx.x;
    if (e >= E) return;
    const int src = ei[e], dst = ei[(long)E + e];
    const float a0 = ea[e * 2], a1 = ea[e * 2 + 1];
    const float* vs = v + (long)src * HCC;
    __builtin_prefetch(vs, 0, 0);
#pragma unroll
    for (int hh = 0; hh < HEADS; ++hh) {
        const float a = att[e * HEADS + hh] /
                        fmaxf(den[(long)dst * HEADS + hh], 1e-16f);
#pragma unroll
        for (int c = 0; c < CH; ++c) {
            const int hc = hh * CH + c;
            const float ev = a0 * Wes[hc] + a1 * Wes[HCC + hc];
            atomicAdd(&acc[(long)dst * HCC + hc], (vs[hc] + ev) * a);
        }
    }
}

// ---------------------------------------------------------------------------
// node: h = ELU(acc1 + s1)
// ---------------------------------------------------------------------------
__global__ void node_elu_kernel(const float* __restrict__ acc,
                                const float* __restrict__ s1,
                                float* __restrict__ h, int N) {
    long i = (long)blockIdx.x * blockDim.x + threadIdx.x;
    if (i >= (long)N * HCC) return;
    const float t = acc[i] + s1[i];
    h[i] = t > 0.f ? t : (__expf(t) - 1.f);
}

// ---------------------------------------------------------------------------
// conv2 edge passes (1 head, 7 channels, no edge features)
// ---------------------------------------------------------------------------
__global__ void edge2_logits_kernel(const float* __restrict__ q,
                                    const float* __restrict__ k,
                                    const int* __restrict__ ei, int E,
                                    float* __restrict__ att,
                                    float* __restrict__ amax) {
    long e = (long)blockIdx.x * blockDim.x + threadIdx.x;
    if (e >= E) return;
    const int src = ei[e], dst = ei[(long)E + e];
    const float* qd = q + (long)dst * OUTF;
    const float* ks = k + (long)src * OUTF;
    float acc = 0.f;
#pragma unroll
    for (int c = 0; c < OUTF; ++c) acc += qd[c] * ks[c];
    acc *= 0.3779644730092272f;                 // 1/sqrt(7)
    att[e] = acc;
    atomicMaxF(&amax[dst], acc);
}

__global__ void edge2_exp_kernel(float* __restrict__ att,
                                 const float* __restrict__ amax,
                                 const int* __restrict__ ei, int E,
                                 float* __restrict__ den) {
    long e = (long)blockIdx.x * blockDim.x + threadIdx.x;
    if (e >= E) return;
    const int dst = ei[(long)E + e];
    const float ex = __expf(att[e] - amax[dst]);
    att[e] = ex;
    atomicAdd(&den[dst], ex);
}

__global__ void edge2_accum_kernel(const float* __restrict__ v,
                                   const float* __restrict__ att,
                                   const float* __restrict__ den,
                                   const int* __restrict__ ei, int E,
                                   float* __restrict__ acc) {
    long e = (long)blockIdx.x * blockDim.x + threadIdx.x;
    if (e >= E) return;
    const int src = ei[e], dst = ei[(long)E + e];
    const float a = att[e] / fmaxf(den[dst], 1e-16f);
    const float* vs = v + (long)src * OUTF;
    __builtin_prefetch(vs, 0, 0);
#pragma unroll
    for (int c = 0; c < OUTF; ++c)
        atomicAdd(&acc[(long)dst * OUTF + c], vs[c] * a);
}

// ---------------------------------------------------------------------------
// final: out = acc2 + s2
// ---------------------------------------------------------------------------
__global__ void final_out_kernel(const float* __restrict__ acc,
                                 const float* __restrict__ s2,
                                 float* __restrict__ out, int N) {
    long i = (long)blockIdx.x * blockDim.x + threadIdx.x;
    if (i < (long)N * OUTF) out[i] = acc[i] + s2[i];
}

// ---------------------------------------------------------------------------
extern "C" void kernel_launch(void* const* d_in, const int* in_sizes, int n_in,
                              void* d_out, int out_size, void* d_ws, size_t ws_size,
                              hipStream_t stream) {
    const float* x   = (const float*)d_in[0];
    const float* ea  = (const float*)d_in[1];
    const int*   ei  = (const int*)  d_in[2];
    const float* Wq1 = (const float*)d_in[3];  const float* bq1 = (const float*)d_in[4];
    const float* Wk1 = (const float*)d_in[5];  const float* bk1 = (const float*)d_in[6];
    const float* Wv1 = (const float*)d_in[7];  const float* bv1 = (const float*)d_in[8];
    const float* We1 = (const float*)d_in[9];
    const float* Ws1 = (const float*)d_in[10]; const float* bs1 = (const float*)d_in[11];
    const float* Wq2 = (const float*)d_in[12]; const float* bq2 = (const float*)d_in[13];
    const float* Wk2 = (const float*)d_in[14]; const float* bk2 = (const float*)d_in[15];
    const float* Wv2 = (const float*)d_in[16]; const float* bv2 = (const float*)d_in[17];
    const float* Ws2 = (const float*)d_in[18]; const float* bs2 = (const float*)d_in[19];

    const int N = in_sizes[0] / IN_F;           // 50000
    const int E = in_sizes[1] / 2;              // 800000

    const size_t N64 = (size_t)N * HCC;
    const size_t N7  = (size_t)N * OUTF;
    const size_t N4  = (size_t)N * HEADS;

    float* p     = (float*)d_ws;
    float* q1    = p; p += N64;
    float* k1    = p; p += N64;
    float* v1    = p; p += N64;
    float* s1    = p; p += N64;
    float* att1  = p; p += (size_t)E * HEADS;   // alpha, then exp(alpha) in place
    float* amax1 = p; p += N4;
    float* den1  = p; p += N4;
    float* acc1  = p; p += N64;
    float* hbuf  = p; p += N64;
    float* q2    = p; p += N7;
    float* k2    = p; p += N7;
    float* v2    = p; p += N7;
    float* s2    = p; p += N7;
    float* att2  = p; p += (size_t)E;
    float* amax2 = p; p += N;
    float* den2  = p; p += N;
    float* acc2  = p; p += N7;

    // f16 packed-weight regions, 64-byte aligned
    uintptr_t up = ((uintptr_t)p + 63) & ~(uintptr_t)63;
    _Float16* pack1 = (_Float16*)up;            // 4*8192 halfs = 64 KB
    _Float16* pack2 = pack1 + 4 * 8192;         // 4*1024 halfs = 8 KB

    float* out = (float*)d_out;
    const int TB = 256;
    const int gInit  = (int)((N64 + TB - 1) / TB);
    const int gRows  = (N + 15) / 16;
    const int gEdge  = (E + TB - 1) / TB;
    const int gNode  = (int)((N64 + TB - 1) / TB);
    const int gFinal = (int)((N7 + TB - 1) / TB);

    init_ws_kernel<<<gInit, TB, 0, stream>>>(amax1, den1, acc1, amax2, den2, acc2, N);
    pack_w1_kernel<<<dim3(32, 4), TB, 0, stream>>>(Wq1, Wk1, Wv1, Ws1, pack1);
    pack_w2_kernel<<<dim3(4, 4),  TB, 0, stream>>>(Wq2, Wk2, Wv2, Ws2, pack2);

    gemm_conv1_kernel<<<gRows, TB, 0, stream>>>(x, pack1, bq1, bk1, bv1, bs1,
                                                q1, k1, v1, s1, N);

    edge1_logits_kernel<<<gEdge, TB, 0, stream>>>(q1, k1, ea, We1, ei, E, att1, amax1);
    edge1_exp_kernel   <<<gEdge, TB, 0, stream>>>(att1, amax1, ei, E, den1);
    edge1_accum_kernel <<<gEdge, TB, 0, stream>>>(v1, ea, We1, att1, den1, ei, E, acc1);

    node_elu_kernel<<<gNode, TB, 0, stream>>>(acc1, s1, hbuf, N);

    gemm_conv2_kernel<<<gRows, TB, 0, stream>>>(hbuf, pack2, bq2, bk2, bv2, bs2,
                                                q2, k2, v2, s2, N);

    edge2_logits_kernel<<<gEdge, TB, 0, stream>>>(q2, k2, ei, E, att2, amax2);
    edge2_exp_kernel   <<<gEdge, TB, 0, stream>>>(att2, amax2, ei, E, den2);
    edge2_accum_kernel <<<gEdge, TB, 0, stream>>>(v2, att2, den2, ei, E, acc2);

    final_out_kernel<<<gFinal, TB, 0, stream>>>(acc2, s2, out, N);
}
